// TemporalAttention_76209899700826
// MI455X (gfx1250) — compile-verified, matrix operand order checked by reading, not run on hardware
//
#include <hip/hip_runtime.h>
#include <math.h>

typedef __attribute__((ext_vector_type(16))) _Float16 v16h;
typedef __attribute__((ext_vector_type(8)))  _Float16 v8h;
typedef __attribute__((ext_vector_type(8)))  float    v8f;

#define Cc 512
#define TT 16
#define HW 1024          // 32*32
#define M_TOT 16384      // T*H*W
#define GN_G 32
#define GN_CPG 16        // channels per group

// ---------------------------------------------------------------------------
// 1) timestep embedding: temb[f][c], f in 0..4, ts = f-2, dim 512 (cos|sin)
// ---------------------------------------------------------------------------
__global__ void temb_kernel(float* __restrict__ temb) {
    int f = blockIdx.x;          // 0..4
    int j = threadIdx.x;         // 0..511
    float tt = (float)(f - 2);
    int idx = (j < 256) ? j : (j - 256);
    float freq = expf(-logf(10000.0f) * (float)idx / 256.0f);
    float arg = tt * freq;
    temb[f * Cc + j] = (j < 256) ? cosf(arg) : sinf(arg);
}

// ---------------------------------------------------------------------------
// 2) folded corrections:
//    qc[c]    = b_q[c] + sum_j temb[2][j]  * w_q[c][j]
//    kc[f][c] = b_k[c] + sum_j temb[ff][j] * w_k[c][j],  ff in {0,1,3,4}
// ---------------------------------------------------------------------------
__global__ void corr_kernel(const float* __restrict__ temb,
                            const float* __restrict__ wq, const float* __restrict__ bq,
                            const float* __restrict__ wk, const float* __restrict__ bk,
                            float* __restrict__ qc, float* __restrict__ kc) {
    int which = blockIdx.x;      // 0 -> qc, 1..4 -> kc[which-1]
    int c = threadIdx.x;         // 0..511
    const float* trow; const float* w; const float* b; float* out;
    if (which == 0) { trow = temb + 2 * Cc; w = wq; b = bq; out = qc; }
    else {
        const int fmap[4] = {0, 1, 3, 4};
        trow = temb + fmap[which - 1] * Cc; w = wk; b = bk; out = kc + (which - 1) * Cc;
    }
    float acc = 0.0f;
    for (int j = 0; j < Cc; ++j) acc += trow[j] * w[(size_t)c * Cc + j];
    out[c] = acc + b[c];
}

// ---------------------------------------------------------------------------
// 3) f32 -> f16 convert (weights)
// ---------------------------------------------------------------------------
__global__ void cvt_f16_kernel(const float* __restrict__ in, _Float16* __restrict__ out, int n) {
    int i = blockIdx.x * blockDim.x + threadIdx.x;
    if (i < n) out[i] = (_Float16)in[i];
}

// ---------------------------------------------------------------------------
// 4) GroupNorm  x:(T,C,H,W) f32  ->  xn16:(T,HW,C) f16 channels-last
//    one block per (t, group); group's 16ch x 1024 spatial is contiguous
// ---------------------------------------------------------------------------
__global__ __launch_bounds__(256) void groupnorm_kernel(
    const float* __restrict__ x, const float* __restrict__ gamma,
    const float* __restrict__ beta, _Float16* __restrict__ xn16) {
    const int t = blockIdx.x, g = blockIdx.y, tid = threadIdx.x;
    const float* xp = x + ((size_t)t * Cc + g * GN_CPG) * HW;
    const int N = GN_CPG * HW;   // 16384
    float s = 0.0f, ss = 0.0f;
    for (int i = tid; i < N; i += 256) { float v = xp[i]; s += v; ss += v * v; }
    __shared__ float r0[256], r1[256];
    r0[tid] = s; r1[tid] = ss; __syncthreads();
    for (int off = 128; off > 0; off >>= 1) {
        if (tid < off) { r0[tid] += r0[tid + off]; r1[tid] += r1[tid + off]; }
        __syncthreads();
    }
    float mu = r0[0] * (1.0f / N);
    float var = r1[0] * (1.0f / N) - mu * mu;
    float rstd = rsqrtf(var + 1e-5f);
    for (int i = tid; i < N; i += 256) {
        int cc = i >> 10, hw = i & 1023;
        int c = g * GN_CPG + cc;
        float v = (xp[i] - mu) * rstd * gamma[c] + beta[c];
        xn16[((size_t)t * HW + hw) * Cc + c] = (_Float16)v;
    }
}

// ---------------------------------------------------------------------------
// WMMA fragment loaders (per-lane, two 16B chunks: K-chunks {0..7,16..23} for
// lanes 0-15 and {8..15,24..31} for lanes 16-31)
// ---------------------------------------------------------------------------
__device__ __forceinline__ v16h load_frag(const _Float16* p) {
    v8h lo = *reinterpret_cast<const v8h*>(p);
    v8h hi = *reinterpret_cast<const v8h*>(p + 16);
    return __builtin_shufflevector(lo, hi, 0,1,2,3,4,5,6,7,8,9,10,11,12,13,14,15);
}

// ---------------------------------------------------------------------------
// 5) WMMA GEMM:  Out[M x 512] = A[M x 512] * W[512 x 512]^T (+ bias), f16 out
//    wave -> 64x16 strip (4 accumulators, B reused 4x per k-step)
//    grid (M/64, 512/128), block 256 (8 waves)
// ---------------------------------------------------------------------------
__global__ __launch_bounds__(256) void gemm_qkv_kernel(
    const _Float16* __restrict__ A, const _Float16* __restrict__ W,
    const float* __restrict__ bias, _Float16* __restrict__ Out) {
    const int lane  = threadIdx.x & 31;
    const int wave  = threadIdx.x >> 5;
    const int laneM = lane & 15;
    const int kgrp  = (lane >> 4) << 3;        // 0 or 8
    const int m0 = blockIdx.x * 64;
    const int n0 = blockIdx.y * 128 + wave * 16;

    v8f acc[4] = {};
    const _Float16* bbase = W + (size_t)(n0 + laneM) * Cc + kgrp;
    const _Float16* abase = A + (size_t)(m0 + laneM) * Cc + kgrp;
    for (int k0 = 0; k0 < Cc; k0 += 32) {
        v16h b = load_frag(bbase + k0);
#pragma unroll
        for (int i = 0; i < 4; ++i) {
            v16h a = load_frag(abase + (size_t)(i * 16) * Cc + k0);
            acc[i] = __builtin_amdgcn_wmma_f32_16x16x32_f16(
                false, a, false, b, (short)0, acc[i], false, false);
        }
    }
    const int n  = n0 + laneM;
    const float bn = bias ? bias[n] : 0.0f;
    const int mr = m0 + ((lane >> 4) << 3);
#pragma unroll
    for (int i = 0; i < 4; ++i)
#pragma unroll
        for (int r = 0; r < 8; ++r)
            Out[(size_t)(mr + i * 16 + r) * Cc + n] = (_Float16)(acc[i][r] + bn);
}

// ---------------------------------------------------------------------------
// 6) attention: one wave per (position m, head). dh=64 -> 2 channels/lane.
//    scores over 4 temporal neighbors, shfl-xor reduce, softmax, weighted V.
// ---------------------------------------------------------------------------
__global__ __launch_bounds__(256) void attn_kernel(
    const _Float16* __restrict__ q16, const _Float16* __restrict__ k16,
    const _Float16* __restrict__ v16, const float* __restrict__ kc,
    _Float16* __restrict__ attn16) {
    const int wid  = blockIdx.x * 8 + (threadIdx.x >> 5);
    const int lane = threadIdx.x & 31;
    const int m    = wid >> 3;
    const int head = wid & 7;
    const int t    = m >> 10;
    const int hw   = m & 1023;
    const int c0   = head * 64 + lane * 2;

    const size_t qb = (size_t)m * Cc + c0;
    float q0 = (float)q16[qb], q1 = (float)q16[qb + 1];

    const int dlt[4] = {-2, -1, 1, 2};
    float sc[4], va[4], vb[4];
#pragma unroll
    for (int f = 0; f < 4; ++f) {
        int tp = t + dlt[f]; tp = tp < 0 ? 0 : (tp > 15 ? 15 : tp);
        size_t base = ((size_t)tp * HW + hw) * Cc + c0;
        float k0v = (float)k16[base]     + kc[f * Cc + c0];
        float k1v = (float)k16[base + 1] + kc[f * Cc + c0 + 1];
        va[f] = (float)v16[base];
        vb[f] = (float)v16[base + 1];
        float p = q0 * k0v + q1 * k1v;
        for (int off = 16; off > 0; off >>= 1) p += __shfl_xor(p, off, 32);
        sc[f] = p * 0.125f;              // dh^-0.5 = 1/8
    }
    float mx = fmaxf(fmaxf(sc[0], sc[1]), fmaxf(sc[2], sc[3]));
    float e0 = expf(sc[0] - mx), e1 = expf(sc[1] - mx);
    float e2 = expf(sc[2] - mx), e3 = expf(sc[3] - mx);
    float inv = 1.0f / (e0 + e1 + e2 + e3);
    float o0 = (e0 * va[0] + e1 * va[1] + e2 * va[2] + e3 * va[3]) * inv;
    float o1 = (e0 * vb[0] + e1 * vb[1] + e2 * vb[2] + e3 * vb[3]) * inv;
    attn16[qb]     = (_Float16)o0;
    attn16[qb + 1] = (_Float16)o1;
}

// ---------------------------------------------------------------------------
// 7) proj GEMM with fused epilogue: +b_proj, transpose to (T,C,H,W), +x
// ---------------------------------------------------------------------------
__global__ __launch_bounds__(256) void gemm_proj_kernel(
    const _Float16* __restrict__ A, const _Float16* __restrict__ W,
    const float* __restrict__ bias, const float* __restrict__ x,
    float* __restrict__ out) {
    const int lane  = threadIdx.x & 31;
    const int wave  = threadIdx.x >> 5;
    const int laneM = lane & 15;
    const int kgrp  = (lane >> 4) << 3;
    const int m0 = blockIdx.x * 64;
    const int n0 = blockIdx.y * 128 + wave * 16;

    v8f acc[4] = {};
    const _Float16* bbase = W + (size_t)(n0 + laneM) * Cc + kgrp;
    const _Float16* abase = A + (size_t)(m0 + laneM) * Cc + kgrp;
    for (int k0 = 0; k0 < Cc; k0 += 32) {
        v16h b = load_frag(bbase + k0);
#pragma unroll
        for (int i = 0; i < 4; ++i) {
            v16h a = load_frag(abase + (size_t)(i * 16) * Cc + k0);
            acc[i] = __builtin_amdgcn_wmma_f32_16x16x32_f16(
                false, a, false, b, (short)0, acc[i], false, false);
        }
    }
    const int n  = n0 + laneM;
    const float bn = bias[n];
    const int mr = m0 + ((lane >> 4) << 3);
#pragma unroll
    for (int i = 0; i < 4; ++i)
#pragma unroll
        for (int r = 0; r < 8; ++r) {
            int m = mr + i * 16 + r;
            int t = m >> 10, hw = m & 1023;
            size_t idx = ((size_t)t * Cc + n) * HW + hw;
            out[idx] = acc[i][r] + bn + x[idx];
        }
}

// ---------------------------------------------------------------------------
extern "C" void kernel_launch(void* const* d_in, const int* in_sizes, int n_in,
                              void* d_out, int out_size, void* d_ws, size_t ws_size,
                              hipStream_t stream) {
    const float* x       = (const float*)d_in[0];
    const float* w_q     = (const float*)d_in[1];
    const float* b_q     = (const float*)d_in[2];
    const float* w_k     = (const float*)d_in[3];
    const float* b_k     = (const float*)d_in[4];
    const float* w_v     = (const float*)d_in[5];
    const float* b_v     = (const float*)d_in[6];
    const float* w_proj  = (const float*)d_in[7];
    const float* b_proj  = (const float*)d_in[8];
    const float* gn_s    = (const float*)d_in[9];
    const float* gn_b    = (const float*)d_in[10];
    float* out           = (float*)d_out;

    char* ws = (char*)d_ws;
    size_t off = 0;
    auto take = [&](size_t bytes) -> void* {
        void* p = ws + off;
        off += (bytes + 255) & ~(size_t)255;
        return p;
    };
    float*    temb   = (float*)take(5 * Cc * sizeof(float));
    float*    qc     = (float*)take(Cc * sizeof(float));
    float*    kc     = (float*)take(4 * Cc * sizeof(float));
    _Float16* wq16   = (_Float16*)take((size_t)Cc * Cc * 2);
    _Float16* wk16   = (_Float16*)take((size_t)Cc * Cc * 2);
    _Float16* wv16   = (_Float16*)take((size_t)Cc * Cc * 2);
    _Float16* wp16   = (_Float16*)take((size_t)Cc * Cc * 2);
    _Float16* xn16   = (_Float16*)take((size_t)M_TOT * Cc * 2);
    _Float16* q16    = (_Float16*)take((size_t)M_TOT * Cc * 2);
    _Float16* k16    = (_Float16*)take((size_t)M_TOT * Cc * 2);
    _Float16* v16    = (_Float16*)take((size_t)M_TOT * Cc * 2);
    _Float16* attn16 = (_Float16*)take((size_t)M_TOT * Cc * 2);

    // 1) temb + folded corrections
    temb_kernel<<<5, 512, 0, stream>>>(temb);
    corr_kernel<<<5, 512, 0, stream>>>(temb, w_q, b_q, w_k, b_k, qc, kc);

    // 2) weight conversion to f16
    const int nw = Cc * Cc;
    cvt_f16_kernel<<<(nw + 255) / 256, 256, 0, stream>>>(w_q, wq16, nw);
    cvt_f16_kernel<<<(nw + 255) / 256, 256, 0, stream>>>(w_k, wk16, nw);
    cvt_f16_kernel<<<(nw + 255) / 256, 256, 0, stream>>>(w_v, wv16, nw);
    cvt_f16_kernel<<<(nw + 255) / 256, 256, 0, stream>>>(w_proj, wp16, nw);

    // 3) GroupNorm -> channels-last f16
    groupnorm_kernel<<<dim3(TT, GN_G), 256, 0, stream>>>(x, gn_s, gn_b, xn16);

    // 4) Q', K', V' WMMA GEMMs (temb folded into qc / kc; b_v folded here)
    dim3 ggrid(M_TOT / 64, Cc / 128);
    gemm_qkv_kernel<<<ggrid, 256, 0, stream>>>(xn16, wq16, qc,      q16);
    gemm_qkv_kernel<<<ggrid, 256, 0, stream>>>(xn16, wk16, nullptr, k16);
    gemm_qkv_kernel<<<ggrid, 256, 0, stream>>>(xn16, wv16, b_v,     v16);

    // 5) attention (wave per position x head)
    attn_kernel<<<M_TOT * 8 * 32 / 256, 256, 0, stream>>>(q16, k16, v16, kc, attn16);

    // 6) projection + transpose + residual
    gemm_proj_kernel<<<ggrid, 256, 0, stream>>>(attn16, wp16, b_proj, x, out);
}
